// LorentzHyperGraphEmbed_73031623901527
// MI455X (gfx1250) — matrix operand order repaired and checked by reading, not compiled, not run on hardware
//
#include <hip/hip_runtime.h>
#include <hip/hip_bf16.h>
#include <math.h>

typedef __attribute__((ext_vector_type(16))) __bf16 v16bf;
typedef __attribute__((ext_vector_type(8)))  float  v8f;
typedef __attribute__((ext_vector_type(4)))  unsigned uint4v;
typedef __attribute__((ext_vector_type(8)))  int  int8v;
typedef __attribute__((ext_vector_type(4)))  int  int4v;

constexpr long Dm  = 256;
constexpr long NDi = 8192;
constexpr long NPRi= 6144;
constexpr long NMi = 2048;
constexpr long NTi = NDi + NPRi + NMi;   // 16384
constexpr long Ei  = 8192;

#define USE_ASYNC_LDS 1
#define USE_TDM 1

#if USE_TDM && __has_builtin(__builtin_amdgcn_tensor_load_to_lds)
#define TDM_OK 1
#else
#define TDM_OK 0
#endif

// ---------------------------------------------------------------------------
// fp32 -> bf16 (round-to-nearest-even)
// ---------------------------------------------------------------------------
__global__ void k_f32_to_bf16(const float* __restrict__ in,
                              unsigned short* __restrict__ out, long n) {
  long i = (long)blockIdx.x * blockDim.x + threadIdx.x;
  if (i < n) {
    unsigned u = __float_as_uint(in[i]);
    out[i] = (unsigned short)((u + 0x7FFFu + ((u >> 16) & 1u)) >> 16);
  }
}

// wave32 + cross-wave row reduction helper (block = 256 threads, 8 waves)
__device__ __forceinline__ float row_reduce(float part, float* wred, int c) {
#pragma unroll
  for (int off = 16; off > 0; off >>= 1) part += __shfl_xor(part, off, 32);
  if ((c & 31) == 0) wred[c >> 5] = part;
  __syncthreads();
  float s = wred[0] + wred[1] + wred[2] + wred[3] +
            wred[4] + wred[5] + wred[6] + wred[7];
  return s;
}

// ---------------------------------------------------------------------------
// lorentz_linear row post-process
// ---------------------------------------------------------------------------
__global__ __launch_bounds__(256)
void k_lorentz_norm(const float* __restrict__ lin, const float* __restrict__ ls,
                    float* __restrict__ out) {
  __shared__ float wred[8];
  __shared__ float s_time;
  int row = blockIdx.x, c = threadIdx.x;
  float v = lin[(long)row * Dm + c];
  if (c == 0) s_time = 1.0f / (1.0f + expf(-v)) * expf(ls[0]) + 1.1f;
  float sum = row_reduce((c == 0) ? 0.0f : v * v, wred, c);
  float t = s_time;
  float scale = sqrtf((t * t - 1.0f) / fmaxf(sum, 1e-8f));
  out[(long)row * Dm + c] = (c == 0) ? t : v * scale;
}

// ---------------------------------------------------------------------------
// lorentz_agg normalize
// ---------------------------------------------------------------------------
__global__ __launch_bounds__(256)
void k_agg_norm(const float* __restrict__ sup, float* __restrict__ out) {
  __shared__ float wred[8];
  int row = blockIdx.x, c = threadIdx.x;
  float v = sup[(long)row * Dm + c];
  float d2 = row_reduce((c == 0) ? v * v : -v * v, wred, c);
  out[(long)row * Dm + c] = v / sqrtf(fmaxf(fabsf(d2), 1e-8f));
}

__global__ void k_gate(const float* __restrict__ lin, const float* __restrict__ e,
                       float* __restrict__ out, long n) {
  long i = (long)blockIdx.x * blockDim.x + threadIdx.x;
  if (i < n) out[i] = e[i] / (1.0f + expf(-lin[i]));
}

// ---------------------------------------------------------------------------
// async 32B global -> LDS copy (GLOBAL_LOAD_ASYNC_TO_LDS_B128 x2).
// INST_OFFSET applies to BOTH LDS and global address (ISA 10.4.4).
// ---------------------------------------------------------------------------
__device__ __forceinline__ void async_copy32(const unsigned short* g,
                                             unsigned short* l) {
#if USE_ASYNC_LDS
  unsigned lds = (unsigned)(unsigned long long)(void*)l;   // low 32b = LDS offset
  unsigned long long ga = (unsigned long long)g;
  asm volatile("global_load_async_to_lds_b128 %0, %1, off\n\t"
               "global_load_async_to_lds_b128 %0, %1, off offset:16"
               :: "v"(lds), "v"(ga) : "memory");
#else
  union { uint4 q; unsigned short s[8]; } q0, q1;
  const uint4* gq = (const uint4*)g;
  q0.q = gq[0]; q1.q = gq[1];
  uint4* lq = (uint4*)l;
  lq[0] = q0.q; lq[1] = q1.q;
#endif
}

__device__ __forceinline__ void async_wait_all() {
#if USE_ASYNC_LDS
  asm volatile("s_wait_asynccnt 0" ::: "memory");
#endif
}

// ---------------------------------------------------------------------------
// TDM: DMA a 128x32-bf16 tile (row stride `rowstride` elements) into LDS with
// padding 16dw + 12dw -> LDS pitch 112B == LDT*2.  D# per ISA ch.8:
//  g0 = {count=1 | ... , lds_addr, global_addr[31:0], ga[56:32]|type=2}
//  g1 = {mask0|data_size=2B|pad_en|pad_int=16dw|pad_amt=12dw, dims/strides}
// ---------------------------------------------------------------------------
__device__ __forceinline__ void tdm_load_tile_a(const unsigned short* g,
                                                unsigned short* lds,
                                                int rowstride) {
#if TDM_OK
  unsigned long long ga = (unsigned long long)g;
  uint4v g0;
  g0[0] = 1u;                                                  // count=1
  g0[1] = (unsigned)(unsigned long long)(void*)lds;            // lds_addr
  g0[2] = (unsigned)(ga & 0xffffffffull);                      // ga[31:0]
  g0[3] = (unsigned)(((ga >> 32) & 0x01ffffffull) | 0x80000000u); // ga[56:32]|type=2
  unsigned td0 = (unsigned)rowstride;      // tensor_dim0 / dim0_stride
  unsigned td1 = 0x00100000u;              // generous tensor_dim1 (no OOB clip)
  int8v g1;
  g1[0] = (int)((1u << 16) | (1u << 20) | (3u << 22) | (11u << 25));
  g1[1] = (int)(td0 << 16);                // tensor_dim0[15:0]
  g1[2] = (int)((td0 >> 16) | (td1 << 16));// tensor_dim0[31:16] | tensor_dim1[15:0]
  g1[3] = (int)((td1 >> 16) | (32u << 16));// tensor_dim1[31:16] | tile_dim0=32
  g1[4] = (int)128;                        // tile_dim1=128, tile_dim2=0
  g1[5] = (int)td0;                        // tensor_dim0_stride[31:0]
  g1[6] = 0;
  g1[7] = 0;
  int4v gz = {0, 0, 0, 0};
#if __clang_major__ >= 23
  int8v gz8 = {0, 0, 0, 0, 0, 0, 0, 0};
  __builtin_amdgcn_tensor_load_to_lds(g0, g1, gz, gz, gz8, 0);
#else
  __builtin_amdgcn_tensor_load_to_lds(g0, g1, gz, gz, 0);
#endif
#endif
}

__device__ __forceinline__ void tensor_wait_all() {
#if TDM_OK
#if __has_builtin(__builtin_amdgcn_s_wait_tensorcnt)
  __builtin_amdgcn_s_wait_tensorcnt(0);
#else
  asm volatile("s_wait_tensorcnt 0" ::: "memory");
#endif
#endif
}

// ---------------------------------------------------------------------------
// Tiled bf16 WMMA GEMM:  C[M,N] = op(A)[M,K] * op(B)[K,N] (+bias[n])
//   transA=1 -> A(m,k) = Amem[k*lda + m]       (H^T @ x)
//   transB=1 -> B(k,n) = Bmem[n*ldb + k]       (x @ W^T)
// 128x128 block tile, 8 waves (4M x 2N), wave = 32x64 = 2x4 WMMA tiles,
// K-step 32 (v_wmma_f32_16x16x32_bf16). Double-buffered LDS.
// Staging: !transA A-tile -> TDM (thread 0); transB B-tile -> async-to-LDS;
// transpose tiles -> VGPR scatter.  LDS pitch 56 bf16 = 112B (16B-aligned,
// bank-conflict-free: 28*m mod 64 distinct for m=0..15).
// ---------------------------------------------------------------------------
#define BM 128
#define BN 128
#define BK 32
#define LDT 56

union FragB16 { v16bf v; unsigned u[8]; };

__global__ __launch_bounds__(256)
void k_gemm_bf16_wmma(const unsigned short* __restrict__ A,
                      const unsigned short* __restrict__ B,
                      const float* __restrict__ bias,
                      float* __restrict__ C,
                      int K, int lda, int ldb, int ldc,
                      int transA, int transB) {
  __shared__ unsigned short As[2][BM * LDT];   // As[m][k]
  __shared__ unsigned short Bs[2][BN * LDT];   // Bs[n][k] (transposed tile)

  const int tid  = threadIdx.x;
  const int lane = tid & 31;
  const int wave = tid >> 5;
  const int wm   = wave >> 1;
  const int wn   = wave & 1;
  const int half = lane >> 4;
  const int l16  = lane & 15;
  const int m0 = blockIdx.x * BM;
  const int n0 = blockIdx.y * BN;

  v8f acc[2][4];
#pragma unroll
  for (int i = 0; i < 2; ++i)
#pragma unroll
    for (int j = 0; j < 4; ++j)
#pragma unroll
      for (int r = 0; r < 8; ++r) acc[i][j][r] = 0.0f;

  auto stage = [&](int sb, int k0) {
    // ---- A tile -> As[sb][m][k] ----
    if (!transA) {
#if TDM_OK
      if (tid == 0)
        tdm_load_tile_a(A + (size_t)m0 * lda + k0, &As[sb][0], lda);
#else
      {
        int r = tid >> 1, cb = (tid & 1) * 16;
        async_copy32(A + (size_t)(m0 + r) * lda + k0 + cb,
                     &As[sb][r * LDT + cb]);
      }
#endif
    } else {
      int kk = tid >> 3, mb = (tid & 7) * 16;
      const unsigned short* gp = A + (size_t)(k0 + kk) * lda + m0 + mb;
      if (k0 + BK < K) __builtin_prefetch(gp + (size_t)BK * lda, 0, 1);
      union { uint4 q; unsigned short s[8]; } q0, q1;
      q0.q = ((const uint4*)gp)[0];
      q1.q = ((const uint4*)gp)[1];
      unsigned short* d = &As[sb][mb * LDT + kk];
#pragma unroll
      for (int j = 0; j < 8; ++j) d[j * LDT] = q0.s[j];
#pragma unroll
      for (int j = 0; j < 8; ++j) d[(8 + j) * LDT] = q1.s[j];
    }
    // ---- B tile -> Bs[sb][n][k] ----
    if (!transB) {
      int kk = tid >> 3, nb = (tid & 7) * 16;
      const unsigned short* gp = B + (size_t)(k0 + kk) * ldb + n0 + nb;
      if (k0 + BK < K) __builtin_prefetch(gp + (size_t)BK * ldb, 0, 1);
      union { uint4 q; unsigned short s[8]; } q0, q1;
      q0.q = ((const uint4*)gp)[0];
      q1.q = ((const uint4*)gp)[1];
      unsigned short* d = &Bs[sb][nb * LDT + kk];
#pragma unroll
      for (int j = 0; j < 8; ++j) d[j * LDT] = q0.s[j];
#pragma unroll
      for (int j = 0; j < 8; ++j) d[(8 + j) * LDT] = q1.s[j];
    } else {
      int n = tid >> 1, kb = (tid & 1) * 16;
      async_copy32(B + (size_t)(n0 + n) * ldb + k0 + kb,
                   &Bs[sb][n * LDT + kb]);
    }
  };

  stage(0, 0);
  int buf = 0;
  for (int k0 = 0; k0 < K; k0 += BK) {
    tensor_wait_all();                     // wave0: TDM into `buf` complete
    async_wait_all();                      // async stores into `buf` complete
    __syncthreads();
    if (k0 + BK < K) stage(buf ^ 1, k0 + BK);   // overlap next tile with WMMAs

    FragB16 af[2], bfr[4];
#pragma unroll
    for (int mt = 0; mt < 2; ++mt) {
      int m = wm * 32 + mt * 16 + l16;
#pragma unroll
      for (int v = 0; v < 8; ++v) {
        int kc = (v < 4 ? 2 * v : 16 + 2 * (v - 4)) + half * 8;
        af[mt].u[v] = *(const unsigned*)(&As[buf][m * LDT + kc]);
      }
    }
#pragma unroll
    for (int nt = 0; nt < 4; ++nt) {
      int n = wn * 64 + nt * 16 + l16;
#pragma unroll
      for (int v = 0; v < 8; ++v) {
        int kc = (v < 4 ? 2 * v : 16 + 2 * (v - 4)) + half * 8;
        bfr[nt].u[v] = *(const unsigned*)(&Bs[buf][n * LDT + kc]);
      }
    }
#pragma unroll
    for (int mt = 0; mt < 2; ++mt)
#pragma unroll
      for (int nt = 0; nt < 4; ++nt)
        acc[mt][nt] = __builtin_amdgcn_wmma_f32_16x16x32_bf16(
            false, af[mt].v, false, bfr[nt].v, (short)0, acc[mt][nt], false, false);

    buf ^= 1;
  }

  // ---- epilogue ----
#pragma unroll
  for (int nt = 0; nt < 4; ++nt) {
    int n = n0 + wn * 64 + nt * 16 + l16;
    float bv = bias ? bias[n] : 0.0f;
#pragma unroll
    for (int mt = 0; mt < 2; ++mt) {
      int mbase = m0 + wm * 32 + mt * 16 + half * 8;
      v8f a = acc[mt][nt];
#pragma unroll
      for (int r = 0; r < 8; ++r)
        C[(size_t)(mbase + r) * ldc + n] = a[r] + bv;
    }
  }
}

// ---------------------------------------------------------------------------
extern "C" void kernel_launch(void* const* d_in, const int* in_sizes, int n_in,
                              void* d_out, int out_size, void* d_ws, size_t ws_size,
                              hipStream_t stream) {
  const float* diseases = (const float*)d_in[0];
  const float* pros     = (const float*)d_in[1];
  const float* meds     = (const float*)d_in[2];
  const float* H        = (const float*)d_in[3];
  const float* Amat     = (const float*)d_in[4];
  const float* ddiA     = (const float*)d_in[5];
  const float* Wh  = (const float*)d_in[6];
  const float* bh  = (const float*)d_in[7];
  const float* sh  = (const float*)d_in[8];
  const float* Wl  = (const float*)d_in[9];
  const float* bl  = (const float*)d_in[10];
  const float* sl  = (const float*)d_in[11];
  const float* Whg = (const float*)d_in[12];
  const float* bhg = (const float*)d_in[13];
  const float* shg = (const float*)d_in[14];
  const float* Wg  = (const float*)d_in[15];
  const float* bg  = (const float*)d_in[16];
  const float* sg  = (const float*)d_in[17];
  const float* Wd  = (const float*)d_in[18];
  const float* bd  = (const float*)d_in[19];
  const float* sd  = (const float*)d_in[20];

  char* ws = (char*)d_ws;
  unsigned short* Hb = (unsigned short*)(ws);                    // 256 MB
  unsigned short* Ab = (unsigned short*)(ws + 268435456LL);      // 128 MB
  unsigned short* Db = (unsigned short*)(ws + 402653184LL);      //   8 MB
  unsigned short* Xb = (unsigned short*)(ws + 411041792LL);      //   8 MB
  unsigned short* Wb = (unsigned short*)(ws + 419430400LL);      // 128 KB
  float* Ef = (float*)(ws + 419561472LL);                        //  16 MB
  float* T0 = (float*)(ws + 436338688LL);
  float* T1 = (float*)(ws + 453115904LL);
  float* T2 = (float*)(ws + 469893120LL);
  float* T3 = (float*)(ws + 486670336LL);                        // l_embed

  float* out        = (float*)d_out;
  float* out_hyper  = out;                  // (8192,256)
  float* out_linear = out + 2097152;        // (8192,256)
  float* out_hembed = out + 4194304;        // (16384,256)
  float* out_ddi    = out + 8388608;        // (2048,256)

  auto to_bf16 = [&](const float* in, unsigned short* o, long n) {
    k_f32_to_bf16<<<dim3((unsigned)((n + 255) / 256)), dim3(256), 0, stream>>>(in, o, n);
  };
  auto gemm = [&](const unsigned short* A_, int lda_, int tA,
                  const unsigned short* B_, int ldb_, int tB,
                  const float* bias_, float* C_, int M_, int N_, int K_) {
    dim3 g((unsigned)(M_ / BM), (unsigned)(N_ / BN));
    k_gemm_bf16_wmma<<<g, 256, 0, stream>>>(A_, B_, bias_, C_, K_, lda_, ldb_, N_, tA, tB);
  };

  to_bf16(H,    Hb, NTi * Ei);
  to_bf16(Amat, Ab, Ei * Ei);
  to_bf16(ddiA, Db, NMi * NMi);

  hipMemcpyAsync(Ef,                     diseases, (size_t)NDi  * Dm * 4, hipMemcpyDeviceToDevice, stream);
  hipMemcpyAsync(Ef + NDi * Dm,          pros,     (size_t)NPRi * Dm * 4, hipMemcpyDeviceToDevice, stream);
  hipMemcpyAsync(Ef + (NDi + NPRi) * Dm, meds,     (size_t)NMi  * Dm * 4, hipMemcpyDeviceToDevice, stream);

  // ---- DDI path ----
  for (int i = 0; i < 2; ++i) {
    to_bf16(Wd + (size_t)i * Dm * Dm, Wb, Dm * Dm);
    to_bf16(i == 0 ? meds : T0, Xb, NMi * Dm);
    gemm(Xb, Dm, 0, Wb, Dm, 1, bd + i * Dm, T1, NMi, Dm, Dm);      // x @ Wd^T + b
    k_lorentz_norm<<<(unsigned)NMi, 256, 0, stream>>>(T1, sd + i, T2);
    to_bf16(T2, Xb, NMi * Dm);
    gemm(Db, NMi, 0, Xb, Dm, 0, nullptr, T1, NMi, Dm, NMi);        // ddi_A @ .
    k_agg_norm<<<(unsigned)NMi, 256, 0, stream>>>(T1, i == 1 ? out_ddi : T0);
  }

  // ---- gated embeddings ----
  to_bf16(Ef, Xb, NTi * Dm);
  to_bf16(Wh, Wb, Dm * Dm);
  gemm(Xb, Dm, 0, Wb, Dm, 1, bh, T1, NTi, Dm, Dm);
  k_lorentz_norm<<<(unsigned)NTi, 256, 0, stream>>>(T1, sh, T2);
  k_gate<<<(unsigned)(NTi * Dm / 256), 256, 0, stream>>>(T2, Ef, T0, NTi * Dm); // h_embed
  to_bf16(Wl, Wb, Dm * Dm);
  gemm(Xb, Dm, 0, Wb, Dm, 1, bl, T1, NTi, Dm, Dm);
  k_lorentz_norm<<<(unsigned)NTi, 256, 0, stream>>>(T1, sl, T2);
  k_gate<<<(unsigned)(NTi * Dm / 256), 256, 0, stream>>>(T2, Ef, T3, NTi * Dm); // l_embed

  // ---- hypergraph loop ----
  for (int i = 0; i < 2; ++i) {
    to_bf16(Whg + (size_t)i * Dm * Dm, Wb, Dm * Dm);
    to_bf16(T0, Xb, NTi * Dm);
    gemm(Xb, Dm, 0, Wb, Dm, 1, bhg + i * Dm, T1, NTi, Dm, Dm);
    k_lorentz_norm<<<(unsigned)NTi, 256, 0, stream>>>(T1, shg + i, T2); // hh
    to_bf16(T2, Xb, NTi * Dm);
    gemm(Hb, Ei, 1, Xb, Dm, 0, nullptr, T1, Ei, Dm, NTi);          // H^T @ hh
    k_agg_norm<<<(unsigned)Ei, 256, 0, stream>>>(T1, T2);          // hh2
    to_bf16(T2, Xb, Ei * Dm);
    gemm(Hb, Ei, 0, Xb, Dm, 0, nullptr, T1, NTi, Dm, Ei);          // H @ hh2
    k_agg_norm<<<(unsigned)NTi, 256, 0, stream>>>(T1, i == 1 ? out_hembed : T0);
  }

  // hyper_rep = agg(h_embed, H^T)
  to_bf16(out_hembed, Xb, NTi * Dm);
  gemm(Hb, Ei, 1, Xb, Dm, 0, nullptr, T1, Ei, Dm, NTi);
  k_agg_norm<<<(unsigned)Ei, 256, 0, stream>>>(T1, out_hyper);

  // ---- linear path ----
  to_bf16(T3, Xb, NTi * Dm);
  gemm(Hb, Ei, 1, Xb, Dm, 0, nullptr, T1, Ei, Dm, NTi);
  k_agg_norm<<<(unsigned)Ei, 256, 0, stream>>>(T1, T0);
  for (int i = 0; i < 2; ++i) {
    to_bf16(Wg + (size_t)i * Dm * Dm, Wb, Dm * Dm);
    to_bf16(T0, Xb, Ei * Dm);
    gemm(Xb, Dm, 0, Wb, Dm, 1, bg + i * Dm, T1, Ei, Dm, Dm);
    k_lorentz_norm<<<(unsigned)Ei, 256, 0, stream>>>(T1, sg + i, T2);
    to_bf16(T2, Xb, Ei * Dm);
    gemm(Ab, Ei, 0, Xb, Dm, 0, nullptr, T1, Ei, Dm, Ei);           // A @ .
    k_agg_norm<<<(unsigned)Ei, 256, 0, stream>>>(T1, i == 1 ? out_linear : T0);
  }

  (void)in_sizes; (void)n_in; (void)out_size; (void)ws_size;
}
